// AE_gnnrnn_57002805953277
// MI455X (gfx1250) — matrix operand order repaired and checked by reading, not compiled, not run on hardware
//
#include <hip/hip_runtime.h>
#include <hip/hip_bf16.h>

// ---------------------------------------------------------------------------
// Problem constants (from the reference)
// ---------------------------------------------------------------------------
#define VN   4000     // vocab
#define DIN  64
#define HH   13       // hidden
#define NN   512      // nodes
#define LL   64       // seq len
#define EE   16384    // edges
#define G4   52       // 4*H
#define H2   26       // 2*H
#define NL   (NN*LL)  // 32768
#define KPAD 32       // padded K for the big WMMA GEMM
#define MT_PER_WAVE 8 // M-tiles processed per wave in the output GEMM

typedef __attribute__((ext_vector_type(16))) _Float16 v16h;
typedef __attribute__((ext_vector_type(8)))  _Float16 v8h;
typedef __attribute__((ext_vector_type(8)))  float    v8f;

__device__ __forceinline__ float sigf(float x) { return 1.0f / (1.0f + __expf(-x)); }

// ---------------------------------------------------------------------------
// Workspace layout (bytes, 256-aligned)
// ---------------------------------------------------------------------------
constexpr size_t AL(size_t x) { return (x + 255) & ~size_t(255); }
constexpr size_t OFF_XEMB = 0;                                     // [NL,64] f32
constexpr size_t OFF_XWF  = AL(OFF_XEMB + (size_t)NL * DIN * 4);   // [NL,52] f32
constexpr size_t OFF_XWB  = AL(OFF_XWF  + (size_t)NL * G4 * 4);    // [NL,52] f32
constexpr size_t OFF_HF   = AL(OFF_XWB  + (size_t)NL * G4 * 4);    // [N,13]
constexpr size_t OFF_CF   = AL(OFF_HF   + (size_t)NN * HH * 4);
constexpr size_t OFF_HB   = AL(OFF_CF   + (size_t)NN * HH * 4);
constexpr size_t OFF_CB   = AL(OFF_HB   + (size_t)NN * HH * 4);
constexpr size_t OFF_STH  = AL(OFF_CB   + (size_t)NN * HH * 4);    // state_h [N,26]
constexpr size_t OFF_STC  = AL(OFF_STH  + (size_t)NN * H2 * 4);    // state_c [N,26]
constexpr size_t OFF_DEG  = AL(OFF_STC  + (size_t)NN * H2 * 4);    // [N]
constexpr size_t OFF_DINV = AL(OFF_DEG  + (size_t)NN * 4);         // [N]
constexpr size_t OFF_GT0  = AL(OFF_DINV + (size_t)NN * 4);         // [N,32] scratch (messages)
constexpr size_t OFF_GT1  = AL(OFF_GT0  + (size_t)NN * 32 * 4);    // [N,32] scratch (agg)
constexpr size_t OFF_GT2  = AL(OFF_GT1  + (size_t)NN * 32 * 4);    // [N,32] scratch (xn)
constexpr size_t OFF_SH   = AL(OFF_GT2  + (size_t)NN * 32 * 4);    // [N,26] decoder h0
constexpr size_t OFF_SC   = AL(OFF_SH   + (size_t)NN * H2 * 4);    // [N,26] decoder c0
constexpr size_t OFF_D16  = AL(OFF_SC   + (size_t)NN * H2 * 4);    // [NL,32] f16 dec_out (K-padded)
constexpr size_t OFF_BP   = AL(OFF_D16  + (size_t)NL * KPAD * 2);  // [V,32] f16 out_W (K-padded)
// total ~24.9 MB

// ---------------------------------------------------------------------------
// 1) Embedding gather: Xemb[nl][d] = emb[x[nl]][d]
// ---------------------------------------------------------------------------
__global__ void k_embed(const int* __restrict__ x, const float* __restrict__ emb,
                        float* __restrict__ xe) {
  int id = blockIdx.x * blockDim.x + threadIdx.x;
  if (id >= NL * DIN) return;
  int nl = id >> 6, d = id & 63;
  int tok = x[nl];
  xe[id] = emb[(size_t)tok * DIN + d];
}

// ---------------------------------------------------------------------------
// 2) Encoder input GEMM: out[nl][g] = sum_d Xemb[nl][d] * Wih[g][d]
//    (small, precision-sensitive feed into the recurrence -> f32 scalar)
// ---------------------------------------------------------------------------
__global__ void k_xw(const float* __restrict__ xe, const float* __restrict__ Wih,
                     float* __restrict__ out) {
  __shared__ float sx[DIN];
  int nl = blockIdx.x, t = threadIdx.x;     // blockDim = 64
  sx[t] = xe[(size_t)nl * DIN + t];
  __syncthreads();
  if (t < G4) {
    float s = 0.f;
#pragma unroll
    for (int d = 0; d < DIN; ++d) s += Wih[t * DIN + d] * sx[d];
    out[(size_t)nl * G4 + t] = s;
  }
}

// ---------------------------------------------------------------------------
// 3) Recurrent encoder: one thread per node, 64 sequential LSTM steps, H=13
// ---------------------------------------------------------------------------
__global__ void k_encoder(const float* __restrict__ xw, const float* __restrict__ Whh,
                          const float* __restrict__ bih, const float* __restrict__ bhh,
                          float* __restrict__ hout, float* __restrict__ cout, int rev) {
  __shared__ float sW[G4 * HH];
  __shared__ float sb[G4];
  int t = threadIdx.x;                      // blockDim = 128
  for (int i = t; i < G4 * HH; i += 128) sW[i] = Whh[i];
  if (t < G4) sb[t] = bih[t] + bhh[t];
  __syncthreads();
  int n = blockIdx.x * 128 + t;             // grid.x = 4 -> 512 nodes
  float h[HH], c[HH], hn[HH];
#pragma unroll
  for (int j = 0; j < HH; ++j) { h[j] = 0.f; c[j] = 0.f; }
  for (int s = 0; s < LL; ++s) {
    int l = rev ? (LL - 1 - s) : s;
    const float* g = xw + ((size_t)n * LL + l) * G4;
#pragma unroll
    for (int j = 0; j < HH; ++j) {
      float ii = g[j]        + sb[j];
      float ff = g[HH + j]   + sb[HH + j];
      float gg = g[2*HH + j] + sb[2*HH + j];
      float oo = g[3*HH + j] + sb[3*HH + j];
#pragma unroll
      for (int k = 0; k < HH; ++k) {
        float hk = h[k];
        ii += sW[(j)        * HH + k] * hk;
        ff += sW[(HH + j)   * HH + k] * hk;
        gg += sW[(2*HH + j) * HH + k] * hk;
        oo += sW[(3*HH + j) * HH + k] * hk;
      }
      float cc = sigf(ff) * c[j] + sigf(ii) * tanhf(gg);
      c[j] = cc;
      hn[j] = sigf(oo) * tanhf(cc);
    }
#pragma unroll
    for (int j = 0; j < HH; ++j) h[j] = hn[j];
  }
#pragma unroll
  for (int j = 0; j < HH; ++j) { hout[n * HH + j] = h[j]; cout[n * HH + j] = c[j]; }
}

// ---------------------------------------------------------------------------
// 4) Projections: state = cat(h_f,h_b) @ projW.T + b   (26x26, per node)
// ---------------------------------------------------------------------------
__global__ void k_proj(const float* hf, const float* hb, const float* cf, const float* cb,
                       const float* p1W, const float* p1b, const float* p2W, const float* p2b,
                       float* sth, float* stc) {
  __shared__ float ch[H2], cc[H2];
  int n = blockIdx.x, t = threadIdx.x;      // blockDim = 64
  if (t < HH) {
    ch[t] = hf[n * HH + t]; ch[HH + t] = hb[n * HH + t];
    cc[t] = cf[n * HH + t]; cc[HH + t] = cb[n * HH + t];
  }
  __syncthreads();
  if (t < H2) {
    float s = p1b[t];
#pragma unroll
    for (int k = 0; k < H2; ++k) s += p1W[t * H2 + k] * ch[k];
    sth[n * H2 + t] = s;
  } else if (t < 2 * H2) {
    int o = t - H2;
    float s = p2b[o];
#pragma unroll
    for (int k = 0; k < H2; ++k) s += p2W[o * H2 + k] * cc[k];
    stc[n * H2 + o] = s;
  }
}

// ---------------------------------------------------------------------------
// 5) GCN helpers
// ---------------------------------------------------------------------------
__global__ void k_deg(const int* __restrict__ dst, float* __restrict__ deg) {
  int i = blockIdx.x * blockDim.x + threadIdx.x;
  if (i >= EE + NN) return;
  int d = (i < EE) ? dst[i] : (i - EE);     // self loops appended
  atomicAdd(&deg[d], 1.0f);
}

__global__ void k_dinv(const float* __restrict__ deg, float* __restrict__ dinv) {
  int i = blockIdx.x * blockDim.x + threadIdx.x;
  if (i >= NN) return;
  float d = deg[i];
  dinv[i] = d > 0.f ? rsqrtf(d) : 0.f;
}

// generic tiny matmul: out[n][o] = (bias?b[o]:0) + sum_k in[n][k]*W[k][o]
__global__ void k_mm(const float* __restrict__ in, const float* __restrict__ W,
                     const float* __restrict__ bias, float* __restrict__ out,
                     int K, int O) {
  int id = blockIdx.x * blockDim.x + threadIdx.x;
  if (id >= NN * O) return;
  int n = id / O, o = id % O;
  float s = bias ? bias[o] : 0.f;
  const float* row = in + (size_t)n * K;
  for (int k = 0; k < K; ++k) s += row[k] * W[k * O + o];
  out[id] = s;
}

// scatter: agg[dst][c] += m[src][c] * dinv[src]*dinv[dst]  (edges + self loops)
__global__ void k_scatter(const float* __restrict__ m, float* __restrict__ agg,
                          const int* __restrict__ src, const int* __restrict__ dst,
                          const float* __restrict__ dinv, int C) {
  int id = blockIdx.x * blockDim.x + threadIdx.x;
  if (id >= (EE + NN) * C) return;
  int e = id / C, c = id % C;
  int s = (e < EE) ? src[e] : (e - EE);
  int d = (e < EE) ? dst[e] : (e - EE);
  float nrm = dinv[s] * dinv[d];
  atomicAdd(&agg[(size_t)d * C + c], m[(size_t)s * C + c] * nrm);
}

__global__ void k_bias_leaky(const float* __restrict__ agg, const float* __restrict__ b,
                             float* __restrict__ out, int C) {
  int id = blockIdx.x * blockDim.x + threadIdx.x;
  if (id >= NN * C) return;
  float v = agg[id] + b[id % C];
  out[id] = v > 0.f ? v : 0.01f * v;
}

// ---------------------------------------------------------------------------
// 6) Decoder: independent single-step bi-LSTM cells; writes f16 A-matrix
//    [NL, 32] with K padded (cols 26..31 = 0) for the WMMA GEMM.
// ---------------------------------------------------------------------------
__global__ void k_decoder(const int* __restrict__ x,
                          const float* __restrict__ sh, const float* __restrict__ sc,
                          const float* Wf, const float* Uf, const float* bif, const float* bhf,
                          const float* Wb, const float* Ub, const float* bib, const float* bhb,
                          _Float16* __restrict__ dec) {
  int nl = blockIdx.x * blockDim.x + threadIdx.x;
  if (nl >= NL) return;
  int n = nl / LL, l = nl % LL;
  float val = (l == 0) ? -1.0f : (float)x[n * LL + (l - 1)];
  _Float16* drow = dec + (size_t)nl * KPAD;
  const float* Wih[2] = { Wf, Wb };
  const float* Whh[2] = { Uf, Ub };
  const float* bi[2]  = { bif, bib };
  const float* bh[2]  = { bhf, bhb };
#pragma unroll
  for (int dir = 0; dir < 2; ++dir) {
    const float* h0 = sh + (size_t)n * H2 + dir * HH;
    const float* c0 = sc + (size_t)n * H2 + dir * HH;
#pragma unroll
    for (int j = 0; j < HH; ++j) {
      float ii = val * Wih[dir][j]        + bi[dir][j]        + bh[dir][j];
      float ff = val * Wih[dir][HH + j]   + bi[dir][HH + j]   + bh[dir][HH + j];
      float gg = val * Wih[dir][2*HH + j] + bi[dir][2*HH + j] + bh[dir][2*HH + j];
      float oo = val * Wih[dir][3*HH + j] + bi[dir][3*HH + j] + bh[dir][3*HH + j];
      float cin = 0.f;
      if (l == 0) {
        cin = c0[j];
        for (int k = 0; k < HH; ++k) {
          float hk = h0[k];
          ii += Whh[dir][(j)        * HH + k] * hk;
          ff += Whh[dir][(HH + j)   * HH + k] * hk;
          gg += Whh[dir][(2*HH + j) * HH + k] * hk;
          oo += Whh[dir][(3*HH + j) * HH + k] * hk;
        }
      }
      float cc = sigf(ff) * cin + sigf(ii) * tanhf(gg);
      float hv = sigf(oo) * tanhf(cc);
      drow[dir * HH + j] = (_Float16)hv;
    }
  }
#pragma unroll
  for (int k = H2; k < KPAD; ++k) drow[k] = (_Float16)0.0f;
}

// ---------------------------------------------------------------------------
// 7) Pack out_W [V,26] f32 -> Bp [V,32] f16 (K zero-padded)
// ---------------------------------------------------------------------------
__global__ void k_prep_b(const float* __restrict__ outW, _Float16* __restrict__ bp) {
  int id = blockIdx.x * blockDim.x + threadIdx.x;
  if (id >= VN * KPAD) return;
  int v = id / KPAD, k = id % KPAD;
  bp[id] = (k < H2) ? (_Float16)outW[(size_t)v * H2 + k] : (_Float16)0.0f;
}

// ---------------------------------------------------------------------------
// 8) Output GEMM: out[32768,4000] = A[32768,26] x out_W.T + out_b
//    Each wave owns one N-tile: loads its B fragment + per-column bias ONCE,
//    then loops over 8 M-tiles (unrolled -> 8 independent v_wmma, overlapping
//    A-loads/stores with WMMA latency). Bias is pre-splatted into the C
//    accumulator (every element of a lane's C belongs to one column), so
//    D = A*B + C_bias needs no epilogue VALU and no WMMA->VALU hazard NOPs.
//    A lane layout (16-bit A 16x32): lane<16 -> row m0+lane, K {0..7,16..23};
//    lane>=16 -> row m0+lane-16, K {8..15,24..31}.  B lane layout (32x16):
//    lane<16 -> col n0+lane, K {0..7,16..23}; lane>=16 -> col, K {8..15,24..31}.
//    C layout: VGPR r -> row m0+r (lane<16) / m0+8+r (lane>=16), col n0+(lane&15).
// ---------------------------------------------------------------------------
__global__ void __launch_bounds__(256) k_outgemm(const _Float16* __restrict__ A,
                                                 const _Float16* __restrict__ Bp,
                                                 const float* __restrict__ bias,
                                                 float* __restrict__ out) {
  const int wave = threadIdx.x >> 5;
  const int lane = threadIdx.x & 31;
  const int tn = blockIdx.y * 8 + wave;           // 250 N-tiles (wave-uniform guard)
  if (tn >= VN / 16) return;
  const int n0 = tn * 16;
  const int lr = lane & 15;
  const bool hi = lane >= 16;

  // B fragment + bias: loaded once, reused for all 8 M-tiles
  const v8h* bpp = (const v8h*)(Bp + ((size_t)(n0 + lr)) * KPAD + (hi ? 8 : 0));
  v8h b0 = bpp[0];
  v8h b1 = bpp[2];
  v16h b;
#pragma unroll
  for (int i = 0; i < 8; ++i) { b[i] = b0[i]; b[8 + i] = b1[i]; }

  const int col = n0 + lr;
  const float bb = bias[col];
  v8f cbias;
#pragma unroll
  for (int r = 0; r < 8; ++r) cbias[r] = bb;

  const int mbase = blockIdx.x * (MT_PER_WAVE * 16);   // grid.x = 2048/8 = 256

#pragma unroll
  for (int mt = 0; mt < MT_PER_WAVE; ++mt) {
    const int m0 = mbase + mt * 16;
    const v8h* ap = (const v8h*)(A + ((size_t)(m0 + lr)) * KPAD + (hi ? 8 : 0));
    v8h a0 = ap[0];          // K base .. base+7
    v8h a1 = ap[2];          // K base+16 .. base+23
    v16h a;
#pragma unroll
    for (int i = 0; i < 8; ++i) { a[i] = a0[i]; a[8 + i] = a1[i]; }

    v8f c = __builtin_amdgcn_wmma_f32_16x16x32_f16(false, a, false, b,
                                                   (short)0, cbias, false, false);

    float* op = out + (size_t)(m0 + (hi ? 8 : 0)) * VN + col;
#pragma unroll
    for (int r = 0; r < 8; ++r) op[(size_t)r * VN] = c[r];
  }
}

// ---------------------------------------------------------------------------
// Host orchestration
// ---------------------------------------------------------------------------
extern "C" void kernel_launch(void* const* d_in, const int* in_sizes, int n_in,
                              void* d_out, int out_size, void* d_ws, size_t ws_size,
                              hipStream_t stream) {
  const int*   x    = (const int*)d_in[0];
  const int*   ei   = (const int*)d_in[1];        // [2,E]: src then dst
  const float* emb  = (const float*)d_in[2];
  const float* eWf  = (const float*)d_in[3];
  const float* eUf  = (const float*)d_in[4];
  const float* ebif = (const float*)d_in[5];
  const float* ebhf = (const float*)d_in[6];
  const float* eWb  = (const float*)d_in[7];
  const float* eUb  = (const float*)d_in[8];
  const float* ebib = (const float*)d_in[9];
  const float* ebhb = (const float*)d_in[10];
  const float* p1W  = (const float*)d_in[11];
  const float* p1b  = (const float*)d_in[12];
  const float* p2W  = (const float*)d_in[13];
  const float* p2b  = (const float*)d_in[14];
  const float* ghW1 = (const float*)d_in[15];
  const float* ghb1 = (const float*)d_in[16];
  const float* ghW2 = (const float*)d_in[17];
  const float* ghb2 = (const float*)d_in[18];
  const float* ghWf = (const float*)d_in[19];
  const float* ghbf = (const float*)d_in[20];
  const float* gcW1 = (const float*)d_in[21];
  const float* gcb1 = (const float*)d_in[22];
  const float* gcW2 = (const float*)d_in[23];
  const float* gcb2 = (const float*)d_in[24];
  const float* gcWf = (const float*)d_in[25];
  const float* gcbf = (const float*)d_in[26];
  const float* dWf  = (const float*)d_in[27];
  const float* dUf  = (const float*)d_in[28];
  const float* dbif = (const float*)d_in[29];
  const float* dbhf = (const float*)d_in[30];
  const float* dWb  = (const float*)d_in[31];
  const float* dUb  = (const float*)d_in[32];
  const float* dbib = (const float*)d_in[33];
  const float* dbhb = (const float*)d_in[34];
  const float* outW = (const float*)d_in[35];
  const float* outb = (const float*)d_in[36];
  float* out = (float*)d_out;

  char* ws = (char*)d_ws;
  float*     XE   = (float*)(ws + OFF_XEMB);
  float*     XWF  = (float*)(ws + OFF_XWF);
  float*     XWB  = (float*)(ws + OFF_XWB);
  float*     HF   = (float*)(ws + OFF_HF);
  float*     CF   = (float*)(ws + OFF_CF);
  float*     HB   = (float*)(ws + OFF_HB);
  float*     CB   = (float*)(ws + OFF_CB);
  float*     STH  = (float*)(ws + OFF_STH);
  float*     STC  = (float*)(ws + OFF_STC);
  float*     DEG  = (float*)(ws + OFF_DEG);
  float*     DINV = (float*)(ws + OFF_DINV);
  float*     GT0  = (float*)(ws + OFF_GT0);
  float*     GT1  = (float*)(ws + OFF_GT1);
  float*     GT2  = (float*)(ws + OFF_GT2);
  float*     SHp  = (float*)(ws + OFF_SH);
  float*     SCp  = (float*)(ws + OFF_SC);
  _Float16*  D16  = (_Float16*)(ws + OFF_D16);
  _Float16*  BP   = (_Float16*)(ws + OFF_BP);

  const int* esrc = ei;
  const int* edst = ei + EE;

  // Embedding + encoder input GEMMs
  k_embed<<<(NL * DIN + 255) / 256, 256, 0, stream>>>(x, emb, XE);
  k_xw<<<NL, 64, 0, stream>>>(XE, eWf, XWF);
  k_xw<<<NL, 64, 0, stream>>>(XE, eWb, XWB);

  // Bidirectional recurrent encoder
  k_encoder<<<NN / 128, 128, 0, stream>>>(XWF, eUf, ebif, ebhf, HF, CF, 0);
  k_encoder<<<NN / 128, 128, 0, stream>>>(XWB, eUb, ebib, ebhb, HB, CB, 1);

  // Projections
  k_proj<<<NN, 64, 0, stream>>>(HF, HB, CF, CB, p1W, p1b, p2W, p2b, STH, STC);

  // GCN normalization
  hipMemsetAsync(DEG, 0, NN * sizeof(float), stream);
  k_deg<<<(EE + NN + 255) / 256, 256, 0, stream>>>(edst, DEG);
  k_dinv<<<(NN + 255) / 256, 256, 0, stream>>>(DEG, DINV);

  // Two GNN stacks: (state, W1,b1,W2,b2,Wfc,bfc) -> out
  auto gnn = [&](const float* xin, const float* W1, const float* b1,
                 const float* W2, const float* b2,
                 const float* Wfc, const float* bfc, float* dst) {
    k_mm<<<(NN * 16 + 255) / 256, 256, 0, stream>>>(xin, W1, nullptr, GT0, H2, 16);
    hipMemsetAsync(GT1, 0, NN * 16 * sizeof(float), stream);
    k_scatter<<<((EE + NN) * 16 + 255) / 256, 256, 0, stream>>>(GT0, GT1, esrc, edst, DINV, 16);
    k_bias_leaky<<<(NN * 16 + 255) / 256, 256, 0, stream>>>(GT1, b1, GT2, 16);
    k_mm<<<(NN * 32 + 255) / 256, 256, 0, stream>>>(GT2, W2, nullptr, GT0, 16, 32);
    hipMemsetAsync(GT1, 0, NN * 32 * sizeof(float), stream);
    k_scatter<<<((EE + NN) * 32 + 255) / 256, 256, 0, stream>>>(GT0, GT1, esrc, edst, DINV, 32);
    k_bias_leaky<<<(NN * 32 + 255) / 256, 256, 0, stream>>>(GT1, b2, GT2, 32);
    k_mm<<<(NN * H2 + 255) / 256, 256, 0, stream>>>(GT2, Wfc, bfc, dst, 32, H2);
  };
  gnn(STH, ghW1, ghb1, ghW2, ghb2, ghWf, ghbf, SHp);
  gnn(STC, gcW1, gcb1, gcW2, gcb2, gcWf, gcbf, SCp);

  // Decoder cells -> f16 A-matrix
  k_decoder<<<(NL + 255) / 256, 256, 0, stream>>>(x, SHp, SCp,
      dWf, dUf, dbif, dbhf, dWb, dUb, dbib, dbhb, D16);

  // Pack weights and run the WMMA output GEMM (dominant cost: 512 MB f32 out)
  k_prep_b<<<(VN * KPAD + 255) / 256, 256, 0, stream>>>(outW, BP);
  k_outgemm<<<dim3((NL / 16) / MT_PER_WAVE, 32), 256, 0, stream>>>(D16, BP, outb, out);
  (void)in_sizes; (void)n_in; (void)out_size; (void)ws_size;
}
